// MultiHeadAttentionBlock_49684181680229
// MI455X (gfx1250) — matrix operand order repaired and checked by reading, not compiled, not run on hardware
//
#include <hip/hip_runtime.h>

// ---------------------------------------------------------------------------
// MultiHeadAttention (reference-faithful: w_q projects Q, K and V; no mask)
// B=2, S=2048, D=1024, H=16, d_k=64.  f16 WMMA, f32 accumulate, flash softmax.
// ---------------------------------------------------------------------------

typedef __attribute__((ext_vector_type(16))) _Float16 v16h;
typedef __attribute__((ext_vector_type(8)))  float    v8f;
typedef __attribute__((ext_vector_type(2)))  _Float16 h2;

#define NB   2
#define NS   2048
#define ND   1024
#define NH   16
#define NDK  64

// ---- WMMA wrapper: v_wmma_f32_16x16x32_f16 (codegen-confirmed builtin) ----
__device__ __forceinline__ v8f wmma_f16(v16h a, v16h b, v8f c) {
  return __builtin_amdgcn_wmma_f32_16x16x32_f16(
      false, a, false, b, (short)0, c, false, false);
}

// A-matrix (16x32 f16) VGPR layout: lane m=lane%16, h=lane/16,
// vgpr v holds K = (v/4)*16 + h*8 + (v&3)*2 , +1
__device__ __forceinline__ v16h load_A_f32(const float* p, int ld, int m, int h) {
  v16h a;
#pragma unroll
  for (int v = 0; v < 8; ++v) {
    int k = ((v >> 2) << 4) + (h << 3) + ((v & 3) << 1);
    const float* q = p + m * ld + k;
    a[2 * v]     = (_Float16)q[0];
    a[2 * v + 1] = (_Float16)q[1];
  }
  return a;
}

__device__ __forceinline__ v16h load_A_f16(const _Float16* p, int ld, int m, int h) {
  v16h a;
#pragma unroll
  for (int v = 0; v < 8; ++v) {
    int k = ((v >> 2) << 4) + (h << 3) + ((v & 3) << 1);
    h2 t = *(const h2*)(p + m * ld + k);
    a[2 * v] = t.x; a[2 * v + 1] = t.y;
  }
  return a;
}

// B-matrix (32x16 f16) layout: lane n=lane%16, h=lane/16,
// vgpr v holds K = (v/4)*16 + h*8 + (v&3)*2 , +1   (same order as A)
// p must point at row-major [n][k] storage (i.e. the transposed operand).
__device__ __forceinline__ v16h load_B_f16(const _Float16* p, int ld, int n, int h) {
  v16h b;
#pragma unroll
  for (int v = 0; v < 8; ++v) {
    int k = ((v >> 2) << 4) + (h << 3) + ((v & 3) << 1);
    h2 t = *(const h2*)(p + n * ld + k);
    b[2 * v] = t.x; b[2 * v + 1] = t.y;
  }
  return b;
}

// row reductions across the 16 lanes holding one C-row (masks stay in-half)
__device__ __forceinline__ float rmax16(float x) {
  x = fmaxf(x, __shfl_xor(x, 1));
  x = fmaxf(x, __shfl_xor(x, 2));
  x = fmaxf(x, __shfl_xor(x, 4));
  x = fmaxf(x, __shfl_xor(x, 8));
  return x;
}
__device__ __forceinline__ float rsum16(float x) {
  x += __shfl_xor(x, 1);
  x += __shfl_xor(x, 2);
  x += __shfl_xor(x, 4);
  x += __shfl_xor(x, 8);
  return x;
}

// ---------------------------------------------------------------------------
// Kernel 1: f32 -> f16 transpose of a 1024x1024 weight (Wt[n][k] = W[k][n])
// ---------------------------------------------------------------------------
__global__ void k_w_transpose(const float* __restrict__ w, _Float16* __restrict__ wt) {
  int idx = blockIdx.x * blockDim.x + threadIdx.x;  // 0 .. 1048575
  int k = idx >> 10, n = idx & 1023;
  wt[n * ND + k] = (_Float16)w[idx];
}

// ---------------------------------------------------------------------------
// Kernel 2: fused QKV projection: P = X @ W_q + b_q  for X in {query,key,value}
// One wave -> one 16x64 tile (16 tokens x one head).  K-loop 1024 in steps 32.
// Q,K stored [bh][s][dk] f16; V stored transposed [bh][dk][s] f16.
// ---------------------------------------------------------------------------
__global__ void __launch_bounds__(256) k_qkv_proj(
    const float* __restrict__ q_in, const float* __restrict__ k_in,
    const float* __restrict__ v_in, const _Float16* __restrict__ wt,
    const float* __restrict__ bias,
    _Float16* __restrict__ q_ws, _Float16* __restrict__ k_ws,
    _Float16* __restrict__ vt_ws) {
  int wave = blockIdx.x * 8 + (threadIdx.x >> 5);
  int lane = threadIdx.x & 31;
  int n16 = lane & 15, hl = lane >> 4;
  int src = wave / (256 * 16);            // 0=q 1=k 2=v (wave-uniform)
  int rem = wave % (256 * 16);
  int rowbase = (rem >> 4) << 4;          // token tile base 0..4080
  int head = rem & 15;
  const float* X = (src == 0) ? q_in : (src == 1) ? k_in : v_in;
  const _Float16* wb = wt + (head * NDK) * ND;

  v8f acc[4] = {};
  for (int kb = 0; kb < ND; kb += 32) {
    v16h a = load_A_f32(X + rowbase * ND + kb, ND, n16, hl);
#pragma unroll
    for (int c = 0; c < 4; ++c) {
      v16h b = load_B_f16(wb + (c * 16) * ND + kb, ND, n16, hl);
      acc[c] = wmma_f16(a, b, acc[c]);
    }
  }

#pragma unroll
  for (int c = 0; c < 4; ++c) {
    int dk  = c * 16 + n16;
    float bv = bias[head * NDK + dk];
#pragma unroll
    for (int v = 0; v < 8; ++v) {
      int row  = rowbase + v + 8 * hl;    // global token 0..4095
      int bidx = row >> 11;
      int s    = row & (NS - 1);
      int bh   = bidx * NH + head;
      _Float16 val = (_Float16)(acc[c][v] + bv);
      if (src == 0)      q_ws[(bh * NS + s) * NDK + dk]  = val;
      else if (src == 1) k_ws[(bh * NS + s) * NDK + dk]  = val;
      else               vt_ws[(bh * NDK + dk) * NS + s] = val;
    }
  }
}

// ---------------------------------------------------------------------------
// Kernel 3: flash attention.  One wave -> one (bh, 16-query tile).
// Per 32-key block: scores (4 WMMA), online softmax, P via LDS, P·V (4 WMMA).
// ---------------------------------------------------------------------------
__global__ void __launch_bounds__(256) k_attention(
    const _Float16* __restrict__ q_ws, const _Float16* __restrict__ k_ws,
    const _Float16* __restrict__ vt_ws, _Float16* __restrict__ x_ws) {
  __shared__ _Float16 lds_p[8][16 * 32];  // 1 KB per wave
  int wslot = threadIdx.x >> 5;
  int wave  = blockIdx.x * 8 + wslot;
  int lane  = threadIdx.x & 31;
  int n16 = lane & 15, hl = lane >> 4;
  int bh    = wave >> 7;                  // 0..31
  int qbase = (wave & 127) << 4;
  _Float16* lp = lds_p[wslot];

  const _Float16* qp = q_ws + (bh * NS + qbase) * NDK;
  v16h a0 = load_A_f16(qp,      NDK, n16, hl);   // d 0..31
  v16h a1 = load_A_f16(qp + 32, NDK, n16, hl);   // d 32..63

  float mi[8], li[8];
  v8f o[4] = {};
#pragma unroll
  for (int v = 0; v < 8; ++v) { mi[v] = -1e30f; li[v] = 0.f; }
  const float scale = 0.125f;  // 1/sqrt(64)

  for (int kb = 0; kb < NS; kb += 32) {
    const _Float16* kp = k_ws + (bh * NS + kb) * NDK;
    v8f s0 = {}, s1 = {};
    {
      v16h b00 = load_B_f16(kp,      NDK, n16, hl);
      v16h b01 = load_B_f16(kp + 32, NDK, n16, hl);
      s0 = wmma_f16(a0, b00, s0);
      s0 = wmma_f16(a1, b01, s0);
      v16h b10 = load_B_f16(kp + 16 * NDK,      NDK, n16, hl);
      v16h b11 = load_B_f16(kp + 16 * NDK + 32, NDK, n16, hl);
      s1 = wmma_f16(a0, b10, s1);
      s1 = wmma_f16(a1, b11, s1);
    }
#pragma unroll
    for (int v = 0; v < 8; ++v) {
      float x0 = s0[v] * scale, x1 = s1[v] * scale;
      float mnew  = fmaxf(mi[v], rmax16(fmaxf(x0, x1)));
      float alpha = __expf(mi[v] - mnew);
      mi[v] = mnew;
      float p0 = __expf(x0 - mnew);
      float p1 = __expf(x1 - mnew);
      li[v] = li[v] * alpha + rsum16(p0 + p1);
#pragma unroll
      for (int c = 0; c < 4; ++c) o[c][v] *= alpha;
      int r = v + 8 * hl;                      // C-layout row
      lp[r * 32 + n16]      = (_Float16)p0;
      lp[r * 32 + n16 + 16] = (_Float16)p1;
    }
    // intra-wave DS RAW: make P visible before re-reading in A layout
    asm volatile("s_wait_dscnt 0" ::: "memory");
    v16h pa = load_A_f16(lp, 32, n16, hl);     // P 16x32, A layout
#pragma unroll
    for (int c = 0; c < 4; ++c) {
      const _Float16* vp = vt_ws + (bh * NDK + c * 16) * NS + kb;
      v16h bv = load_B_f16(vp, NS, n16, hl);   // V^T rows contiguous in k
      o[c] = wmma_f16(pa, bv, o[c]);
    }
  }

  int b = bh >> 4, hh = bh & 15;
#pragma unroll
  for (int v = 0; v < 8; ++v) {
    float inv = 1.f / li[v];
    int tok = b * NS + qbase + v + 8 * hl;
#pragma unroll
    for (int c = 0; c < 4; ++c) {
      int col = hh * NDK + c * 16 + n16;       // merged-head column
      x_ws[tok * ND + col] = (_Float16)(o[c][v] * inv);
    }
  }
}

// ---------------------------------------------------------------------------
// Kernel 4: output projection  out = x @ w_o + b_o   (f32 result)
// ---------------------------------------------------------------------------
__global__ void __launch_bounds__(256) k_out_proj(
    const _Float16* __restrict__ x_ws, const _Float16* __restrict__ wot,
    const float* __restrict__ b_o, float* __restrict__ out) {
  int wave = blockIdx.x * 8 + (threadIdx.x >> 5);
  int lane = threadIdx.x & 31;
  int n16 = lane & 15, hl = lane >> 4;
  int rowbase = (wave >> 4) << 4;
  int cg = wave & 15;                          // 64-column group

  v8f acc[4] = {};
  for (int kb = 0; kb < ND; kb += 32) {
    v16h a = load_A_f16(x_ws + rowbase * ND + kb, ND, n16, hl);
#pragma unroll
    for (int c = 0; c < 4; ++c) {
      v16h b = load_B_f16(wot + (cg * 64 + c * 16) * ND + kb, ND, n16, hl);
      acc[c] = wmma_f16(a, b, acc[c]);
    }
  }
#pragma unroll
  for (int c = 0; c < 4; ++c) {
    int col = cg * 64 + c * 16 + n16;
    float bv = b_o[col];
#pragma unroll
    for (int v = 0; v < 8; ++v) {
      int row = rowbase + v + 8 * hl;
      out[row * ND + col] = acc[c][v] + bv;
    }
  }
}

// ---------------------------------------------------------------------------
// Launch.  Workspace layout (36 MB total):
//   [0,2)MB wq_t | [2,4)MB wo_t | [4,12)MB q | [12,20)MB k | [20,28)MB v^T
//   [28,36)MB x (merged heads, f16)
// ---------------------------------------------------------------------------
extern "C" void kernel_launch(void* const* d_in, const int* in_sizes, int n_in,
                              void* d_out, int out_size, void* d_ws, size_t ws_size,
                              hipStream_t stream) {
  const float* query = (const float*)d_in[0];
  const float* key_  = (const float*)d_in[1];
  const float* value = (const float*)d_in[2];
  const float* w_q   = (const float*)d_in[3];
  const float* b_q   = (const float*)d_in[4];
  const float* w_o   = (const float*)d_in[5];
  const float* b_o   = (const float*)d_in[6];
  float* out = (float*)d_out;

  char* ws = (char*)d_ws;
  _Float16* wq_t  = (_Float16*)(ws);
  _Float16* wo_t  = (_Float16*)(ws + (size_t)(2)  * 1024 * 1024);
  _Float16* q_ws  = (_Float16*)(ws + (size_t)(4)  * 1024 * 1024);
  _Float16* k_ws  = (_Float16*)(ws + (size_t)(12) * 1024 * 1024);
  _Float16* vt_ws = (_Float16*)(ws + (size_t)(20) * 1024 * 1024);
  _Float16* x_ws  = (_Float16*)(ws + (size_t)(28) * 1024 * 1024);

  k_w_transpose<<<4096, 256, 0, stream>>>(w_q, wq_t);
  k_w_transpose<<<4096, 256, 0, stream>>>(w_o, wo_t);
  // 3 matrices x 256 row-tiles x 16 heads = 12288 waves, 8 waves/block
  k_qkv_proj<<<1536, 256, 0, stream>>>(query, key_, value, wq_t, b_q,
                                       q_ws, k_ws, vt_ws);
  // 32 bh x 128 query-tiles = 4096 waves
  k_attention<<<512, 256, 0, stream>>>(q_ws, k_ws, vt_ws, x_ws);
  // 256 row-tiles x 16 col-groups = 4096 waves
  k_out_proj<<<512, 256, 0, stream>>>(x_ws, wo_t, b_o, out);
}
